// WordCharLSTMCRF_56556129354532
// MI455X (gfx1250) — compile-verified, matrix operand order checked by reading, not run on hardware
//
#include <hip/hip_runtime.h>
#include <hip/hip_bf16.h>
#include <math.h>

typedef __bf16 bf16;
typedef __bf16 v16bf __attribute__((ext_vector_type(16)));
typedef float  v8f   __attribute__((ext_vector_type(8)));

// ---------------------------------------------------------------------------
// WMMA fragment loader. For C = A(MxK) @ W^T with W stored (N,K) row-major,
// the A-tile (16x32) and B-tile both use this loader:
//   lanes 0-15 : row r, K = 0..7 (elems 0..7) and K = 16..23 (elems 8..15)
//   lanes 16-31: row r, K = 8..15 and K = 24..31
// (CDNA5 ISA 7.12.2, 16-bit A-matrix layout; B mirrors with rows of W.)
// ---------------------------------------------------------------------------
__device__ __forceinline__ v16bf load_frag(const bf16* p0, int ld) {
  int lane = threadIdx.x & 31;
  int r = lane & 15;
  int hi = lane >> 4;
  const bf16* p = p0 + (size_t)r * ld + hi * 8;
  v16bf v;
#pragma unroll
  for (int e = 0; e < 8; ++e) v[e] = p[e];
#pragma unroll
  for (int e = 0; e < 8; ++e) v[8 + e] = p[16 + e];
  return v;
}

__device__ __forceinline__ void wmma_acc(v8f& acc, v16bf a, v16bf b) {
  acc = __builtin_amdgcn_wmma_f32_16x16x32_bf16(false, a, false, b, (short)0,
                                                acc, false, false);
}

__device__ __forceinline__ float sigmoidf_(float x) { return 1.0f / (1.0f + expf(-x)); }

// ---------------------------------------------------------------------------
// Generic WMMA GEMM: C(MxN) = A(MxK,bf16) @ W^T (W is (N,K) bf16) + bias.
// Block = 128 threads = 4 waves arranged 2x2 -> 32x32 output tile per block.
// TANH_BF16=1: write tanh(val) as bf16 (FF1). Else write f32 (FF2 logits).
// ---------------------------------------------------------------------------
template <int TANH_BF16>
__global__ void gemm_bias_kernel(const bf16* __restrict__ A, const bf16* __restrict__ W,
                                 const float* __restrict__ bias,
                                 float* __restrict__ Cf, bf16* __restrict__ Cb,
                                 int M, int N, int K) {
  int wave = threadIdx.x >> 5;
  int m0 = (blockIdx.y * 2 + (wave >> 1)) * 16;
  int n0 = (blockIdx.x * 2 + (wave & 1)) * 16;
  v8f acc = {};
  for (int k = 0; k < K; k += 32) {
    v16bf a = load_frag(A + (size_t)m0 * K + k, K);
    v16bf b = load_frag(W + (size_t)n0 * K + k, K);
    wmma_acc(acc, a, b);
  }
  int lane = threadIdx.x & 31;
  int col = lane & 15, hi = lane >> 4;
  float bb = bias[n0 + col];
#pragma unroll
  for (int v = 0; v < 8; ++v) {
    int row = m0 + v + hi * 8;
    float val = acc[v] + bb;
    if (TANH_BF16)
      Cb[(size_t)row * N + n0 + col] = (bf16)tanhf(val);
    else
      Cf[(size_t)row * N + n0 + col] = val;
  }
}

// ---------------------------------------------------------------------------
// Fused LSTM step:  z = x_t @ Wih^T + h @ Whh^T + (bih+bhh) ; gate math ; c,h.
// Grid (H/16, rows/16), block 128 = 4 waves; wave g computes gate g's 16x16
// z-tile with WMMA, exchanges through LDS, then 128 threads do the cell.
// Cin/Cout may alias (block touches only its own (row,col) tile of c).
// Hout must ping-pong (other column-blocks read full h rows).
// ---------------------------------------------------------------------------
__global__ void lstm_step_kernel(const bf16* __restrict__ Xt, int Kx,
                                 const bf16* __restrict__ Wih,   // (4H, Kx) bf16
                                 const bf16* __restrict__ Hin,   // (rows, H) bf16
                                 const bf16* __restrict__ Whh,   // (4H, H) bf16
                                 const float* __restrict__ bias, // (4H) = bih+bhh
                                 const float* __restrict__ Cin,
                                 float* __restrict__ Cout,
                                 bf16* __restrict__ Hout,
                                 bf16* __restrict__ HsOut, int ld_hs,
                                 int H) {
  __shared__ float zs[4][16][17];
  int g = threadIdx.x >> 5;          // 0=i 1=f 2=g 3=o  (torch gate order)
  int m0 = blockIdx.y * 16;
  int c0 = blockIdx.x * 16;          // column inside H
  int n0 = g * H + c0;               // column inside 4H gate space
  v8f acc = {};
  for (int k = 0; k < Kx; k += 32) {
    v16bf a = load_frag(Xt + (size_t)m0 * Kx + k, Kx);
    v16bf b = load_frag(Wih + (size_t)n0 * Kx + k, Kx);
    wmma_acc(acc, a, b);
  }
  for (int k = 0; k < H; k += 32) {
    v16bf a = load_frag(Hin + (size_t)m0 * H + k, H);
    v16bf b = load_frag(Whh + (size_t)n0 * H + k, H);
    wmma_acc(acc, a, b);
  }
  int lane = threadIdx.x & 31;
  int col = lane & 15, hi = lane >> 4;
  float bb = bias[n0 + col];
#pragma unroll
  for (int v = 0; v < 8; ++v) zs[g][v + hi * 8][col] = acc[v] + bb;
  __syncthreads();
  for (int i = threadIdx.x; i < 256; i += 128) {
    int r = i >> 4, cc = i & 15;
    float zi = zs[0][r][cc], zf = zs[1][r][cc];
    float zg = zs[2][r][cc], zo = zs[3][r][cc];
    size_t idx = (size_t)(m0 + r) * H + c0 + cc;
    float c = sigmoidf_(zf) * Cin[idx] + sigmoidf_(zi) * tanhf(zg);
    float h = sigmoidf_(zo) * tanhf(c);
    Cout[idx] = c;
    Hout[idx] = (bf16)h;
    if (HsOut) HsOut[(size_t)(m0 + r) * ld_hs + c0 + cc] = (bf16)h;
  }
}

// ---------------------------------------------------------------------------
// Small helper kernels
// ---------------------------------------------------------------------------
__global__ void cvt_pad_kernel(const float* __restrict__ src, bf16* __restrict__ dst,
                               int rows, int ks, int kd) {
  int i = blockIdx.x * blockDim.x + threadIdx.x;
  if (i >= rows * kd) return;
  int c = i % kd, r = i / kd;
  dst[i] = (c < ks) ? (bf16)src[(size_t)r * ks + c] : (bf16)0.0f;
}

__global__ void vec_add_kernel(const float* a, const float* b, float* o, int n) {
  int i = blockIdx.x * blockDim.x + threadIdx.x;
  if (i < n) o[i] = a[i] + b[i];
}

__global__ void fill_u32_kernel(unsigned* p, int n) {
  int i = blockIdx.x * blockDim.x + threadIdx.x;
  if (i < n) p[i] = 0u;
}

// charX[t][n][e] = char_emb[char_inputs[n*W+t]][e], n = b*S+s, written as bf16
__global__ void gather_char_kernel(const int* __restrict__ ci,
                                   const float* __restrict__ emb,
                                   bf16* __restrict__ charX,
                                   int N, int W, int Ec) {
  int i = blockIdx.x * blockDim.x + threadIdx.x;
  if (i >= W * N * Ec) return;
  int e = i % Ec;
  int rest = i / Ec;
  int n = rest % N;
  int t = rest / N;
  int idx = ci[n * W + t];
  charX[i] = (bf16)emb[(size_t)idx * Ec + e];
}

// xw row (s*B+b), cols: [0,E)=word_emb, [E,E+Hc)=char fwd hT, [E+Hc,E+2Hc)=char
// bwd hT (char state rows are n=b*S+s), [E+2Hc]=cap_emb, pad to Dp with zeros.
__global__ void build_xw_kernel(const int* __restrict__ wi, const int* __restrict__ capi,
                                const float* __restrict__ wemb, const float* __restrict__ cemb,
                                const bf16* __restrict__ chf, const bf16* __restrict__ chb,
                                bf16* __restrict__ xw,
                                int S, int B, int E, int Hc, int Dp) {
  int i = blockIdx.x * blockDim.x + threadIdx.x;
  if (i >= S * B * Dp) return;
  int c = i % Dp;
  int row = i / Dp;           // s*B + b
  int s = row / B, b = row % B;
  float v;
  if (c < E)               v = wemb[(size_t)wi[row] * E + c];
  else if (c < E + Hc)     v = (float)chf[(size_t)(b * S + s) * Hc + (c - E)];
  else if (c < E + 2 * Hc) v = (float)chb[(size_t)(b * S + s) * Hc + (c - E - Hc)];
  else if (c == E + 2 * Hc) v = cemb[capi[row]];
  else                     v = 0.0f;
  xw[i] = (bf16)v;
}

// ---------------------------------------------------------------------------
// CRF forward: one wave32 per batch (lane = tag, T=32 exactly fills a wave).
// logsumexp over 'from' via lane shuffles; gold score on lane 0.
// ---------------------------------------------------------------------------
__global__ void crf_forward_kernel(const float* __restrict__ logits,
                                   const int* __restrict__ tags,
                                   const float* __restrict__ trans,
                                   const float* __restrict__ startv,
                                   const float* __restrict__ endv,
                                   float* __restrict__ out_scores, int S, int B) {
  __shared__ float tr[1024];
  for (int i = threadIdx.x; i < 1024; i += blockDim.x) tr[i] = trans[i];
  __syncthreads();
  int lane = threadIdx.x & 31;
  int wave = threadIdx.x >> 5;
  int b = blockIdx.x * 16 + wave;
  float alpha = startv[lane] + logits[(size_t)b * 32 + lane];
  for (int s = 1; s < S; ++s) {
    float m = -INFINITY;
    for (int f = 0; f < 32; ++f) {
      float af = __shfl(alpha, f, 32);
      m = fmaxf(m, af + tr[f * 32 + lane]);
    }
    float sum = 0.0f;
    for (int f = 0; f < 32; ++f) {
      float af = __shfl(alpha, f, 32);
      sum += expf(af + tr[f * 32 + lane] - m);
    }
    alpha = m + logf(sum) + logits[((size_t)s * B + b) * 32 + lane];
  }
  float v = alpha + endv[lane];
  float m = v;
  for (int o = 16; o; o >>= 1) m = fmaxf(m, __shfl_xor(m, o, 32));
  float e = expf(v - m);
  for (int o = 16; o; o >>= 1) e += __shfl_xor(e, o, 32);
  float logZ = m + logf(e);
  if (lane == 0) {
    int prev = tags[b];
    float sc = startv[prev] + endv[tags[(size_t)(S - 1) * B + b]] +
               logits[(size_t)b * 32 + prev];
    for (int s = 1; s < S; ++s) {
      int t = tags[(size_t)s * B + b];
      sc += logits[((size_t)s * B + b) * 32 + t] + tr[prev * 32 + t];
      prev = t;
    }
    out_scores[b] = sc - logZ;
  }
}

// ---------------------------------------------------------------------------
// Viterbi: one wave32 per batch; uint8 backpointers; lane-0 backtrace.
// ---------------------------------------------------------------------------
__global__ void viterbi_kernel(const float* __restrict__ logits,
                               const float* __restrict__ trans,
                               const float* __restrict__ startv,
                               const float* __restrict__ endv,
                               unsigned char* __restrict__ bp,
                               float* __restrict__ out_tags, int S, int B) {
  __shared__ float tr[1024];
  for (int i = threadIdx.x; i < 1024; i += blockDim.x) tr[i] = trans[i];
  __syncthreads();
  int lane = threadIdx.x & 31;
  int wave = threadIdx.x >> 5;
  int b = blockIdx.x * 16 + wave;
  float v = startv[lane] + logits[(size_t)b * 32 + lane];
  for (int s = 1; s < S; ++s) {
    float best = -INFINITY;
    int bf = 0;
    for (int f = 0; f < 32; ++f) {
      float af = __shfl(v, f, 32);
      float sc = af + tr[f * 32 + lane];
      if (sc > best) { best = sc; bf = f; }
    }
    bp[((size_t)(s - 1) * B + b) * 32 + lane] = (unsigned char)bf;
    v = best + logits[((size_t)s * B + b) * 32 + lane];
  }
  float val = v + endv[lane];
  float bm = val;
  int bi = lane;
  for (int o = 16; o; o >>= 1) {
    float om = __shfl_xor(bm, o, 32);
    int oi = __shfl_xor(bi, o, 32);
    if (om > bm || (om == bm && oi < bi)) { bm = om; bi = oi; }
  }
  if (lane == 0) {
    int tag = bi;
    out_tags[(size_t)b * S + (S - 1)] = (float)tag;
    for (int s = S - 2; s >= 0; --s) {
      tag = bp[((size_t)s * B + b) * 32 + tag];
      out_tags[(size_t)b * S + s] = (float)tag;
    }
  }
}

__global__ void finalize_loss_kernel(const float* scores, float* out, int B) {
  if (threadIdx.x == 0 && blockIdx.x == 0) {
    float s = 0.0f;
    for (int b = 0; b < B; ++b) s += scores[b];
    out[0] = -s;
  }
}

// ---------------------------------------------------------------------------
// Host orchestration
// ---------------------------------------------------------------------------
extern "C" void kernel_launch(void* const* d_in, const int* in_sizes, int n_in,
                              void* d_out, int out_size, void* d_ws, size_t ws_size,
                              hipStream_t stream) {
  constexpr int S = 256, B = 64, W = 16, Ec = 64, Hc = 128, Hw = 512;
  constexpr int E = 300, T = 32, D = 557, Dp = 576;
  constexpr int N = B * S;  // 16384 word positions

  const int*   word_inputs = (const int*)d_in[0];
  const int*   char_inputs = (const int*)d_in[1];
  const int*   cap_inputs  = (const int*)d_in[2];
  const int*   tag_inputs  = (const int*)d_in[3];
  const float* word_emb    = (const float*)d_in[4];
  const float* cap_emb     = (const float*)d_in[5];
  const float* char_emb    = (const float*)d_in[6];

  char* base = (char*)d_ws;
  size_t off = 0;
  auto alloc = [&](size_t bytes) -> void* {
    void* p = base + off;
    off += (bytes + 255) & ~(size_t)255;
    return p;
  };

  // bf16 weights (torch (N,K) row-major maps directly to WMMA B operand)
  bf16* cWihF = (bf16*)alloc((size_t)4 * Hc * Ec * 2);
  bf16* cWhhF = (bf16*)alloc((size_t)4 * Hc * Hc * 2);
  bf16* cWihB = (bf16*)alloc((size_t)4 * Hc * Ec * 2);
  bf16* cWhhB = (bf16*)alloc((size_t)4 * Hc * Hc * 2);
  bf16* wWihF = (bf16*)alloc((size_t)4 * Hw * Dp * 2);
  bf16* wWhhF = (bf16*)alloc((size_t)4 * Hw * Hw * 2);
  bf16* wWihB = (bf16*)alloc((size_t)4 * Hw * Dp * 2);
  bf16* wWhhB = (bf16*)alloc((size_t)4 * Hw * Hw * 2);
  bf16* ff1Wb = (bf16*)alloc((size_t)Hw * (2 * Hw) * 2);
  bf16* ff2Wb = (bf16*)alloc((size_t)T * Hw * 2);
  float* bcf = (float*)alloc(4 * Hc * 4);
  float* bcb = (float*)alloc(4 * Hc * 4);
  float* bwf = (float*)alloc(4 * Hw * 4);
  float* bwb = (float*)alloc(4 * Hw * 4);
  // activations / state
  bf16* charX = (bf16*)alloc((size_t)W * N * Ec * 2);
  bf16* hf0 = (bf16*)alloc((size_t)N * Hc * 2);
  bf16* hf1 = (bf16*)alloc((size_t)N * Hc * 2);
  bf16* hb0 = (bf16*)alloc((size_t)N * Hc * 2);
  bf16* hb1 = (bf16*)alloc((size_t)N * Hc * 2);
  float* cfS = (float*)alloc((size_t)N * Hc * 4);
  float* cbS = (float*)alloc((size_t)N * Hc * 4);
  bf16* xw  = (bf16*)alloc((size_t)N * Dp * 2);
  bf16* wh0 = (bf16*)alloc((size_t)B * Hw * 2);
  bf16* wh1 = (bf16*)alloc((size_t)B * Hw * 2);
  float* wcS = (float*)alloc((size_t)B * Hw * 4);
  bf16* enc = (bf16*)alloc((size_t)N * 2 * Hw * 2);
  bf16* hff = (bf16*)alloc((size_t)N * Hw * 2);
  float* logits = (float*)alloc((size_t)N * T * 4);
  unsigned char* bp = (unsigned char*)alloc((size_t)S * B * T);
  float* scores = (float*)alloc(B * 4);
  (void)ws_size; (void)n_in; (void)in_sizes; (void)out_size;

  auto nb = [](int n) { return (n + 255) / 256; };
  // weight conversions (pad word Wih K: 557 -> 576)
  cvt_pad_kernel<<<nb(4 * Hc * Ec), 256, 0, stream>>>((const float*)d_in[7],  cWihF, 4 * Hc, Ec, Ec);
  cvt_pad_kernel<<<nb(4 * Hc * Hc), 256, 0, stream>>>((const float*)d_in[8],  cWhhF, 4 * Hc, Hc, Hc);
  cvt_pad_kernel<<<nb(4 * Hc * Ec), 256, 0, stream>>>((const float*)d_in[11], cWihB, 4 * Hc, Ec, Ec);
  cvt_pad_kernel<<<nb(4 * Hc * Hc), 256, 0, stream>>>((const float*)d_in[12], cWhhB, 4 * Hc, Hc, Hc);
  cvt_pad_kernel<<<nb(4 * Hw * Dp), 256, 0, stream>>>((const float*)d_in[15], wWihF, 4 * Hw, D, Dp);
  cvt_pad_kernel<<<nb(4 * Hw * Hw), 256, 0, stream>>>((const float*)d_in[16], wWhhF, 4 * Hw, Hw, Hw);
  cvt_pad_kernel<<<nb(4 * Hw * Dp), 256, 0, stream>>>((const float*)d_in[19], wWihB, 4 * Hw, D, Dp);
  cvt_pad_kernel<<<nb(4 * Hw * Hw), 256, 0, stream>>>((const float*)d_in[20], wWhhB, 4 * Hw, Hw, Hw);
  cvt_pad_kernel<<<nb(Hw * 2 * Hw), 256, 0, stream>>>((const float*)d_in[23], ff1Wb, Hw, 2 * Hw, 2 * Hw);
  cvt_pad_kernel<<<nb(T * Hw),      256, 0, stream>>>((const float*)d_in[25], ff2Wb, T, Hw, Hw);
  vec_add_kernel<<<nb(4 * Hc), 256, 0, stream>>>((const float*)d_in[9],  (const float*)d_in[10], bcf, 4 * Hc);
  vec_add_kernel<<<nb(4 * Hc), 256, 0, stream>>>((const float*)d_in[13], (const float*)d_in[14], bcb, 4 * Hc);
  vec_add_kernel<<<nb(4 * Hw), 256, 0, stream>>>((const float*)d_in[17], (const float*)d_in[18], bwf, 4 * Hw);
  vec_add_kernel<<<nb(4 * Hw), 256, 0, stream>>>((const float*)d_in[21], (const float*)d_in[22], bwb, 4 * Hw);

  // char embeddings
  gather_char_kernel<<<nb(W * N * Ec), 256, 0, stream>>>(char_inputs, char_emb, charX, N, W, Ec);

  // zero LSTM states
  fill_u32_kernel<<<nb(N * Hc / 2), 256, 0, stream>>>((unsigned*)hf0, N * Hc / 2);
  fill_u32_kernel<<<nb(N * Hc / 2), 256, 0, stream>>>((unsigned*)hb0, N * Hc / 2);
  fill_u32_kernel<<<nb(N * Hc), 256, 0, stream>>>((unsigned*)cfS, N * Hc);
  fill_u32_kernel<<<nb(N * Hc), 256, 0, stream>>>((unsigned*)cbS, N * Hc);

  // char BiLSTM (16 fused WMMA steps each direction; final h ends in *0)
  dim3 cgrid(Hc / 16, N / 16);
  for (int t = 0; t < W; ++t) {
    bf16* hin = (t & 1) ? hf1 : hf0;
    bf16* hout = (t & 1) ? hf0 : hf1;
    lstm_step_kernel<<<cgrid, 128, 0, stream>>>(charX + (size_t)t * N * Ec, Ec,
        cWihF, hin, cWhhF, bcf, cfS, cfS, hout, nullptr, 0, Hc);
  }
  for (int i = 0; i < W; ++i) {
    int t = W - 1 - i;
    bf16* hin = (i & 1) ? hb1 : hb0;
    bf16* hout = (i & 1) ? hb0 : hb1;
    lstm_step_kernel<<<cgrid, 128, 0, stream>>>(charX + (size_t)t * N * Ec, Ec,
        cWihB, hin, cWhhB, bcb, cbS, cbS, hout, nullptr, 0, Hc);
  }

  // build word-LSTM input (S*B, 576): [word_emb | char hT fwd | char hT bwd | cap | pad]
  build_xw_kernel<<<nb(N * Dp), 256, 0, stream>>>(word_inputs, cap_inputs, word_emb,
      cap_emb, hf0, hb0, xw, S, B, E, Hc, Dp);

  // word BiLSTM; hs written directly into packed enc (stride 1024)
  dim3 wgrid(Hw / 16, B / 16);
  fill_u32_kernel<<<nb(B * Hw / 2), 256, 0, stream>>>((unsigned*)wh0, B * Hw / 2);
  fill_u32_kernel<<<nb(B * Hw), 256, 0, stream>>>((unsigned*)wcS, B * Hw);
  for (int t = 0; t < S; ++t) {
    bf16* hin = (t & 1) ? wh1 : wh0;
    bf16* hout = (t & 1) ? wh0 : wh1;
    lstm_step_kernel<<<wgrid, 128, 0, stream>>>(xw + (size_t)t * B * Dp, Dp,
        wWihF, hin, wWhhF, bwf, wcS, wcS, hout,
        enc + (size_t)t * B * 2 * Hw, 2 * Hw, Hw);
  }
  fill_u32_kernel<<<nb(B * Hw / 2), 256, 0, stream>>>((unsigned*)wh0, B * Hw / 2);
  fill_u32_kernel<<<nb(B * Hw), 256, 0, stream>>>((unsigned*)wcS, B * Hw);
  for (int i = 0; i < S; ++i) {
    int t = S - 1 - i;
    bf16* hin = (i & 1) ? wh1 : wh0;
    bf16* hout = (i & 1) ? wh0 : wh1;
    lstm_step_kernel<<<wgrid, 128, 0, stream>>>(xw + (size_t)t * B * Dp, Dp,
        wWihB, hin, wWhhB, bwb, wcS, wcS, hout,
        enc + (size_t)t * B * 2 * Hw + Hw, 2 * Hw, Hw);
  }

  // FF1: tanh(enc @ ff1_W^T + b) -> bf16 ; FF2: logits f32
  gemm_bias_kernel<1><<<dim3(Hw / 32, N / 32), 128, 0, stream>>>(
      enc, ff1Wb, (const float*)d_in[24], nullptr, hff, N, Hw, 2 * Hw);
  gemm_bias_kernel<0><<<dim3(T / 32, N / 32), 128, 0, stream>>>(
      hff, ff2Wb, (const float*)d_in[26], logits, nullptr, N, T, Hw);

  // CRF loss + Viterbi decode
  const float* trans  = (const float*)d_in[27];
  const float* startv = (const float*)d_in[28];
  const float* endv   = (const float*)d_in[29];
  crf_forward_kernel<<<4, 512, 0, stream>>>(logits, tag_inputs, trans, startv, endv,
                                            scores, S, B);
  viterbi_kernel<<<4, 512, 0, stream>>>(logits, trans, startv, endv, bp,
                                        (float*)d_out + 1, S, B);
  finalize_loss_kernel<<<1, 32, 0, stream>>>(scores, (float*)d_out, B);
}
